// SDC_25769803776707
// MI455X (gfx1250) — matrix operand order repaired and checked by reading, not compiled, not run on hardware
//
#include <hip/hip_runtime.h>
#include <hip/hip_bf16.h>
#include <math.h>

// ---------------------------------------------------------------------------
// Problem constants (from reference): B=8, C=64, H=W=192, heads=16, ch=4
// ---------------------------------------------------------------------------
#define BB   8
#define CC   64
#define HH   192
#define WW   192
#define NPIX (HH*WW)            // 36864
#define K9   (CC*9)             // 576 (implicit-GEMM K for 3x3 conv)

typedef __attribute__((ext_vector_type(16))) _Float16 v16h;
typedef __attribute__((ext_vector_type(8)))  _Float16 v8h;
typedef __attribute__((ext_vector_type(8)))  float    v8f;
typedef __attribute__((ext_vector_type(4)))  int      v4i;

#if defined(__has_builtin)
#if __has_builtin(__builtin_amdgcn_global_load_async_to_lds_b128) && \
    __has_builtin(__builtin_amdgcn_s_wait_asynccnt)
#define HAVE_ASYNC_LDS 1
#endif
#endif
#ifndef HAVE_ASYNC_LDS
#define HAVE_ASYNC_LDS 0
#endif

#if HAVE_ASYNC_LDS
// global (AS1) and LDS (AS3) typed pointers for the async-copy builtin
#define ASYNC_GPTR(p) ((__attribute__((address_space(1))) v4i*)(void*)(p))
#define ASYNC_LPTR(p) ((__attribute__((address_space(3))) v4i*)(void*)(p))
#endif

// ---------------------------------------------------------------------------
// Fragment loads (CDNA5 16-bit WMMA layouts, wave32).
// A (16xK): lane m = lane&15; lanes<16 hold K kb+0..15 (same order as B),
//           lanes>=16 K kb+16..31  -> one 32B run via ld_b_contig.
// B (Kx16): lane n = lane&15; lanes<16 hold K kb+0..15, lanes>=16 kb+16..31
//           -> one 32B contiguous run when K is the minor LDS dimension.
// ---------------------------------------------------------------------------
__device__ __forceinline__ v16h ld_b_contig(const _Float16* p) {
  v8h a = *(const v8h*)p;
  v8h b = *(const v8h*)(p + 8);
  return __builtin_shufflevector(a, b, 0, 1, 2, 3, 4, 5, 6, 7,
                                 8, 9, 10, 11, 12, 13, 14, 15);
}
__device__ __forceinline__ v16h ld_a_pair(const _Float16* p) {
  v8h a = *(const v8h*)p;
  v8h b = *(const v8h*)(p + 16);
  return __builtin_shufflevector(a, b, 0, 1, 2, 3, 4, 5, 6, 7,
                                 8, 9, 10, 11, 12, 13, 14, 15);
}

// ---------------------------------------------------------------------------
// 0) Weight prep: 3x3 dense weights -> f16 with K reordered to
//    k' = (ky*3+kx)*64 + ci  (so 32-wide K blocks are channel-contiguous);
//    1x1 weights -> f16 verbatim.
// ---------------------------------------------------------------------------
__global__ __launch_bounds__(256)
void cvt_weights_kernel(const float* __restrict__ w1, const float* __restrict__ w2,
                        const float* __restrict__ wq, const float* __restrict__ wp,
                        _Float16* __restrict__ o1, _Float16* __restrict__ o2,
                        _Float16* __restrict__ oq, _Float16* __restrict__ op) {
  int i = blockIdx.x * 256 + threadIdx.x;
  if (i < CC * K9) {
    int o = i / K9, rem = i - o * K9;
    int ci = rem / 9, r = rem - ci * 9;
    int kp = r * 64 + ci;
    o1[o * K9 + kp] = (_Float16)w1[i];
    o2[o * K9 + kp] = (_Float16)w2[i];
  }
  if (i < 192 * 64) oq[i] = (_Float16)wq[i];
  if (i < 64 * 64)  op[i] = (_Float16)wp[i];
}

// ---------------------------------------------------------------------------
// 1) Dense 3x3 conv (64->64) via implicit-GEMM WMMA, optional fused BN.
//    Block = 256 thr (8 waves): 2 output rows x 64 cols x 64 out-channels.
//    Halo patch staged channel-minor: patch[(ry*66+cx)*72 + ci], 16B-aligned
//    rows so B fragments are two ds_load_b128.
// ---------------------------------------------------------------------------
__global__ __launch_bounds__(256)
void conv3x3_wmma_kernel(const float* __restrict__ in, const _Float16* __restrict__ w,
                         const float* __restrict__ gamma, const float* __restrict__ beta,
                         const float* __restrict__ mean,  const float* __restrict__ var,
                         float* __restrict__ out, int use_bn) {
  __shared__ alignas(16) _Float16 patch[4 * 66 * 72];   // 38016 B

  const int tid = threadIdx.x;
  const int bx  = blockIdx.x;
  const int xt  = bx % (WW / 64);
  const int yb  = (bx / (WW / 64)) % (HH / 2);
  const int b   = bx / ((WW / 64) * (HH / 2));
  const int x0  = xt * 64;
  const int y0  = yb * 2;

  // stage halo patch (rows y0-1 .. y0+2), zero padded, channel-minor
  for (int i = tid; i < CC * 4 * 66; i += 256) {
    int ci  = i / 264;
    int rem = i - ci * 264;
    int ry  = rem / 66;
    int cx  = rem - ry * 66;
    int yy  = y0 + ry - 1;
    int xx  = x0 - 1 + cx;
    float v = 0.f;
    if (yy >= 0 && yy < HH && xx >= 0 && xx < WW)
      v = in[((size_t)(b * CC + ci) * HH + yy) * WW + xx];
    patch[(ry * 66 + cx) * 72 + ci] = (_Float16)v;
  }
  __syncthreads();

  const int lane   = tid & 31;
  const int wave   = tid >> 5;
  const int wr     = wave >> 2;           // output row within block (0/1)
  const int n_base = (wave & 3) * 16;     // column strip
  const int cxb    = n_base + (lane & 15);

  v8f acc[4];
#pragma unroll
  for (int mt = 0; mt < 4; ++mt)
#pragma unroll
    for (int e = 0; e < 8; ++e) acc[mt][e] = 0.f;

  const _Float16* wrow = w + (size_t)(lane & 15) * K9 + ((lane < 16) ? 0 : 16);

  for (int kb = 0; kb < K9; kb += 32) {
    int k0  = kb + ((lane < 16) ? 0 : 16);
    int r   = k0 >> 6;                    // kernel position 0..8
    int ci0 = k0 & 63;
    int ry  = r / 3, rx = r - ry * 3;
    v16h bf = ld_b_contig(&patch[((wr + ry) * 66 + cxb + rx) * 72 + ci0]);
#pragma unroll
    for (int mt = 0; mt < 4; ++mt) {
      v16h af = ld_b_contig(wrow + (size_t)(mt * 16) * K9 + kb);
      acc[mt] = __builtin_amdgcn_wmma_f32_16x16x32_f16(false, af, false, bf,
                                                       (short)0, acc[mt], false, false);
    }
  }

  const int y  = y0 + wr;
  const int xo = x0 + n_base + (lane & 15);
#pragma unroll
  for (int mt = 0; mt < 4; ++mt) {
#pragma unroll
    for (int rr = 0; rr < 8; ++rr) {
      int c = mt * 16 + rr + ((lane >= 16) ? 8 : 0);
      float v = acc[mt][rr];
      if (use_bn) {
        float inv = gamma[c] * rsqrtf(var[c] + 1e-5f);
        v = v * inv + (beta[c] - mean[c] * inv);
      }
      out[((size_t)(b * CC + c) * HH + y) * WW + xo] = v;
    }
  }
}

// ---------------------------------------------------------------------------
// 2) Fused depthwise diff convs: dp = dw(x,xk) * dw(g,gk)^2   (C=64)
// ---------------------------------------------------------------------------
__global__ __launch_bounds__(256)
void diffprod_kernel(const float* __restrict__ x, const float* __restrict__ xk,
                     const float* __restrict__ g, const float* __restrict__ gk,
                     float* __restrict__ dp) {
  size_t idx = (size_t)blockIdx.x * 256 + threadIdx.x;
  if (idx >= (size_t)BB * CC * NPIX) return;
  int xx = idx % WW;
  int yy = (idx / WW) % HH;
  size_t bc = idx / NPIX;
  int c = (int)(bc % CC);
  float xd = 0.f, gd = 0.f;
#pragma unroll
  for (int ky = 0; ky < 3; ++ky)
#pragma unroll
    for (int kx = 0; kx < 3; ++kx) {
      int yi = yy + ky - 1, xi = xx + kx - 1;
      if (yi >= 0 && yi < HH && xi >= 0 && xi < WW) {
        float iv = x[bc * NPIX + (size_t)yi * WW + xi];
        float gv = g[bc * NPIX + (size_t)yi * WW + xi];
        xd += iv * xk[c * 9 + ky * 3 + kx];
        gd += gv * gk[c * 9 + ky * 3 + kx];
      }
    }
  dp[idx] = xd * gd * gd;
}

// ---------------------------------------------------------------------------
// 3) 1x1 conv via WMMA. Cin=64; MT = Cout/16 templated (4 or 12).
//    Block = 256 thr (8 waves) x 128 positions. Activations staged
//    transposed (K minor). f16 weights staged via async global->LDS copy
//    (GLOBAL_LOAD_ASYNC_TO_LDS_B128) when the builtin is available.
// ---------------------------------------------------------------------------
template <int MT>
__global__ __launch_bounds__(256)
void conv1x1_wmma_kernel(const float* __restrict__ in, const _Float16* __restrict__ w,
                         const float* __restrict__ addend, float* __restrict__ out,
                         int add_flag) {
  constexpr int COUT = MT * 16;
  __shared__ alignas(16) _Float16 xlds[128 * 72];    // [col][ci] transposed
  __shared__ alignas(16) _Float16 wlds[COUT * CC];   // [Cout][64] row-major

  const int tid = threadIdx.x;
  const int bx  = blockIdx.x;
  const int b   = bx / (NPIX / 128);
  const int p0  = (bx % (NPIX / 128)) * 128;

#if HAVE_ASYNC_LDS
  for (int i = tid; i < (COUT * CC) / 8; i += 256)
    __builtin_amdgcn_global_load_async_to_lds_b128(
        ASYNC_GPTR(w + i * 8), ASYNC_LPTR(&wlds[i * 8]), 0, 0);
#else
  for (int i = tid; i < COUT * CC; i += 256) wlds[i] = w[i];
#endif

  for (int i = tid; i < CC * 128; i += 256) {
    int col = i & 127, ci = i >> 7;
    xlds[col * 72 + ci] = (_Float16)in[((size_t)b * CC + ci) * NPIX + p0 + col];
  }
#if HAVE_ASYNC_LDS
  __builtin_amdgcn_s_wait_asynccnt(0);
#endif
  __syncthreads();

  const int lane   = tid & 31;
  const int wave   = tid >> 5;
  const int n_base = wave * 16;

  v8f acc[MT];
#pragma unroll
  for (int mt = 0; mt < MT; ++mt)
#pragma unroll
    for (int e = 0; e < 8; ++e) acc[mt][e] = 0.f;

  const _Float16* xrow = &xlds[(n_base + (lane & 15)) * 72];
  const _Float16* wrow = &wlds[(lane & 15) * CC + ((lane < 16) ? 0 : 16)];

#pragma unroll
  for (int kb = 0; kb < CC; kb += 32) {
    v16h bf = ld_b_contig(xrow + kb + ((lane < 16) ? 0 : 16));
#pragma unroll
    for (int mt = 0; mt < MT; ++mt) {
      v16h af = ld_b_contig(wrow + (size_t)(mt * 16) * CC + kb);
      acc[mt] = __builtin_amdgcn_wmma_f32_16x16x32_f16(false, af, false, bf,
                                                       (short)0, acc[mt], false, false);
    }
  }

  const int po = p0 + n_base + (lane & 15);
#pragma unroll
  for (int mt = 0; mt < MT; ++mt) {
#pragma unroll
    for (int rr = 0; rr < 8; ++rr) {
      int c = mt * 16 + rr + ((lane >= 16) ? 8 : 0);
      size_t o = ((size_t)b * COUT + c) * NPIX + po;
      float v = acc[mt][rr];
      if (add_flag) v += addend[o];
      out[o] = v;
    }
  }
}

// ---------------------------------------------------------------------------
// 4) Generic depthwise 3x3 conv (qkv_dw, C=192)
// ---------------------------------------------------------------------------
__global__ __launch_bounds__(256)
void dwconv_kernel(const float* __restrict__ in, const float* __restrict__ kw,
                   float* __restrict__ out, int C) {
  size_t idx = (size_t)blockIdx.x * 256 + threadIdx.x;
  if (idx >= (size_t)BB * C * NPIX) return;
  int xx = idx % WW;
  int yy = (idx / WW) % HH;
  size_t bc = idx / NPIX;
  int c = (int)(bc % C);
  float acc = 0.f;
#pragma unroll
  for (int ky = 0; ky < 3; ++ky)
#pragma unroll
    for (int kx = 0; kx < 3; ++kx) {
      int yi = yy + ky - 1, xi = xx + kx - 1;
      if (yi >= 0 && yi < HH && xi >= 0 && xi < WW)
        acc += in[bc * NPIX + (size_t)yi * WW + xi] * kw[c * 9 + ky * 3 + kx];
    }
  out[idx] = acc;
}

// ---------------------------------------------------------------------------
// 5) Per-(b,head) Gram reductions: sum q^2 (4), sum k^2 (4), q.k^T (16)
// ---------------------------------------------------------------------------
__global__ __launch_bounds__(256)
void gram_kernel(const float* __restrict__ qkv, float* __restrict__ red) {
  int bh = blockIdx.x;
  int b = bh >> 4, head = bh & 15;
  const float* q = qkv + ((size_t)(b * 192 + head * 4)) * NPIX;
  const float* k = q + (size_t)64 * NPIX;

  float qq[4] = {0, 0, 0, 0}, kk[4] = {0, 0, 0, 0}, qk[16];
#pragma unroll
  for (int i = 0; i < 16; ++i) qk[i] = 0.f;

  for (int n = threadIdx.x; n < NPIX; n += 256) {
    float qv[4], kv[4];
#pragma unroll
    for (int c = 0; c < 4; ++c) {
      qv[c] = q[(size_t)c * NPIX + n];
      kv[c] = k[(size_t)c * NPIX + n];
    }
#pragma unroll
    for (int c = 0; c < 4; ++c) { qq[c] += qv[c] * qv[c]; kk[c] += kv[c] * kv[c]; }
#pragma unroll
    for (int c = 0; c < 4; ++c)
#pragma unroll
      for (int d = 0; d < 4; ++d) qk[c * 4 + d] += qv[c] * kv[d];
  }

  __shared__ float s[24];
  if (threadIdx.x < 24) s[threadIdx.x] = 0.f;
  __syncthreads();
#pragma unroll
  for (int i = 0; i < 4; ++i) { atomicAdd(&s[i], qq[i]); atomicAdd(&s[4 + i], kk[i]); }
#pragma unroll
  for (int i = 0; i < 16; ++i) atomicAdd(&s[8 + i], qk[i]);
  __syncthreads();
  if (threadIdx.x < 24) red[bh * 24 + threadIdx.x] = s[threadIdx.x];
}

// ---------------------------------------------------------------------------
// 6) Combined 4x4 mix matrix: attn = Gram/(|q||k|)*temp; W = sum a_i *
//    topk_softmax(attn, k_i), k_i in {2,2,3,3}. One thread per (b,head).
// ---------------------------------------------------------------------------
__global__ __launch_bounds__(128)
void wcomb_kernel(const float* __restrict__ red, const float* __restrict__ temp,
                  const float* __restrict__ a1, const float* __restrict__ a2,
                  const float* __restrict__ a3, const float* __restrict__ a4,
                  float* __restrict__ wcomb) {
  int t = blockIdx.x * blockDim.x + threadIdx.x;
  if (t >= BB * 16) return;
  int head = t & 15;
  const float* r = red + t * 24;

  float attn[16];
#pragma unroll
  for (int c = 0; c < 4; ++c) {
    float nq = fmaxf(sqrtf(r[c]), 1e-12f);
#pragma unroll
    for (int d = 0; d < 4; ++d) {
      float nk = fmaxf(sqrtf(r[4 + d]), 1e-12f);
      attn[c * 4 + d] = (r[8 + c * 4 + d] / (nq * nk)) * temp[head];
    }
  }

  const float as[4] = {a1[0], a2[0], a3[0], a4[0]};
  const int ks[4] = {2, 2, 3, 3};
  float W[16];
#pragma unroll
  for (int i = 0; i < 16; ++i) W[i] = 0.f;

#pragma unroll
  for (int i = 0; i < 4; ++i) {
#pragma unroll
    for (int c = 0; c < 4; ++c) {
      float v0 = attn[c * 4 + 0], v1 = attn[c * 4 + 1];
      float v2 = attn[c * 4 + 2], v3 = attn[c * 4 + 3];
      float tmpv;
#define CSW(a, b) if (a < b) { tmpv = a; a = b; b = tmpv; }
      CSW(v0, v1) CSW(v2, v3) CSW(v0, v2) CSW(v1, v3) CSW(v1, v2)
#undef CSW
      float sorted[4] = {v0, v1, v2, v3};
      float thresh = sorted[ks[i] - 1];
      float mx = -1e30f;
#pragma unroll
      for (int d = 0; d < 4; ++d) {
        float v = attn[c * 4 + d];
        if (v >= thresh && v > mx) mx = v;
      }
      float e[4], ssum = 0.f;
#pragma unroll
      for (int d = 0; d < 4; ++d) {
        float v = attn[c * 4 + d];
        e[d] = (v >= thresh) ? expf(v - mx) : 0.f;
        ssum += e[d];
      }
#pragma unroll
      for (int d = 0; d < 4; ++d) W[c * 4 + d] += as[i] * e[d] / ssum;
    }
  }
#pragma unroll
  for (int i = 0; i < 16; ++i) wcomb[t * 16 + i] = W[i];
}

// ---------------------------------------------------------------------------
// 7) Apply rank-4 mix: out[head*4+c, n] = sum_d W[c][d] * v[d, n]
// ---------------------------------------------------------------------------
__global__ __launch_bounds__(256)
void attnmix_kernel(const float* __restrict__ qkv, const float* __restrict__ wcomb,
                    float* __restrict__ out) {
  size_t idx = (size_t)blockIdx.x * 256 + threadIdx.x;
  if (idx >= (size_t)BB * 16 * NPIX) return;
  int n = (int)(idx % NPIX);
  int bh = (int)(idx / NPIX);
  int b = bh >> 4, head = bh & 15;
  const float* v = qkv + ((size_t)(b * 192 + 128 + head * 4)) * NPIX + n;
  const float* W = wcomb + bh * 16;
  float vv[4];
#pragma unroll
  for (int d = 0; d < 4; ++d) vv[d] = v[(size_t)d * NPIX];
#pragma unroll
  for (int c = 0; c < 4; ++c) {
    float o = W[c * 4 + 0] * vv[0] + W[c * 4 + 1] * vv[1] +
              W[c * 4 + 2] * vv[2] + W[c * 4 + 3] * vv[3];
    out[((size_t)(b * CC + head * 4 + c)) * NPIX + n] = o;
  }
}

// ---------------------------------------------------------------------------
// Host launch
// ---------------------------------------------------------------------------
extern "C" void kernel_launch(void* const* d_in, const int* in_sizes, int n_in,
                              void* d_out, int out_size, void* d_ws, size_t ws_size,
                              hipStream_t stream) {
  (void)in_sizes; (void)n_in; (void)out_size; (void)ws_size;
  const float* x          = (const float*)d_in[0];
  const float* guidance   = (const float*)d_in[1];
  const float* conv1_w    = (const float*)d_in[2];
  const float* bn_gamma   = (const float*)d_in[3];
  const float* bn_beta    = (const float*)d_in[4];
  const float* bn_mean    = (const float*)d_in[5];
  const float* bn_var     = (const float*)d_in[6];
  const float* x_kdiff    = (const float*)d_in[7];
  const float* g_kdiff    = (const float*)d_in[8];
  const float* qkv_w      = (const float*)d_in[9];
  const float* qkv_dw_w   = (const float*)d_in[10];
  const float* proj_w     = (const float*)d_in[11];
  const float* temp       = (const float*)d_in[12];
  const float* a1         = (const float*)d_in[13];
  const float* a2         = (const float*)d_in[14];
  const float* a3         = (const float*)d_in[15];
  const float* a4         = (const float*)d_in[16];
  const float* conv_w     = (const float*)d_in[17];
  float* out = (float*)d_out;

  const size_t NT = (size_t)BB * CC * NPIX;      // 18,874,368
  float* wsf     = (float*)d_ws;
  float* g_buf   = wsf;                          // g; later reused as (dp+attn_out)
  float* dp_buf  = wsf + NT;
  float* bufA    = wsf + 2 * NT;                 // 3*NT: qkv point; later attnmix out
  float* bufB    = wsf + 5 * NT;                 // 3*NT: qkv after depthwise
  float* red     = wsf + 8 * NT;                 // 128*24
  float* wcomb   = red + 4096;                   // 128*16
  _Float16* w1h  = (_Float16*)(wcomb + 4096);    // conv1_w f16 [64][576] (K reordered)
  _Float16* w2h  = w1h + CC * K9;                // conv_w  f16 [64][576] (K reordered)
  _Float16* wqh  = w2h + CC * K9;                // qkv_w   f16 [192][64]
  _Float16* wph  = wqh + 192 * 64;               // proj_w  f16 [64][64]

  const int g3_blocks = (WW / 64) * (HH / 2) * BB;  // 2304
  const int g1_blocks = (NPIX / 128) * BB;          // 2304

  // 0) weights -> f16 (3x3 weights K-reordered)
  cvt_weights_kernel<<<(CC * K9 + 255) / 256, 256, 0, stream>>>(
      conv1_w, conv_w, qkv_w, proj_w, w1h, w2h, wqh, wph);
  // 1) g = BN(conv3x3(guidance))
  conv3x3_wmma_kernel<<<g3_blocks, 256, 0, stream>>>(
      guidance, w1h, bn_gamma, bn_beta, bn_mean, bn_var, g_buf, 1);
  // 2) dp = dw(x)*dw(g)^2
  diffprod_kernel<<<(unsigned)((NT + 255) / 256), 256, 0, stream>>>(
      x, x_kdiff, g_buf, g_kdiff, dp_buf);
  // 3) qkv point conv (64 -> 192)
  conv1x1_wmma_kernel<12><<<g1_blocks, 256, 0, stream>>>(dp_buf, wqh, nullptr, bufA, 0);
  // 4) qkv depthwise 3x3 (192 groups)
  dwconv_kernel<<<(unsigned)(((size_t)BB * 192 * NPIX + 255) / 256), 256, 0, stream>>>(
      bufA, qkv_dw_w, bufB, 192);
  // 5) per-(b,head) Gram reductions
  gram_kernel<<<BB * 16, 256, 0, stream>>>(bufB, red);
  // 6) combined top-k softmax mix matrices
  wcomb_kernel<<<1, 128, 0, stream>>>(red, temp, a1, a2, a3, a4, wcomb);
  // 7) attn = W @ v  (into bufA, reused)
  attnmix_kernel<<<(unsigned)(((size_t)BB * 16 * NPIX + 255) / 256), 256, 0, stream>>>(
      bufB, wcomb, bufA);
  // 8) proj 1x1 + add dp  -> g_buf (reused as sum)
  conv1x1_wmma_kernel<4><<<g1_blocks, 256, 0, stream>>>(bufA, wph, dp_buf, g_buf, 1);
  // 9) final conv3x3 -> out
  conv3x3_wmma_kernel<<<g3_blocks, 256, 0, stream>>>(
      g_buf, w2h, bn_gamma, bn_beta, bn_mean, bn_var, out, 0);
}
